// XoronMoE_87445534147205
// MI455X (gfx1250) — matrix operand order repaired and checked by reading, not compiled, not run on hardware
//
#include <hip/hip_runtime.h>
#include <hip/hip_bf16.h>

// Problem constants (B=2, S=4096, H=1024, E=8, K=2, I=1408)
#define T_TOKENS 8192
#define H_DIM 1024
#define E_NUM 8
#define I_DIM 1408

typedef __attribute__((ext_vector_type(16))) __bf16 v16bf;
typedef __attribute__((ext_vector_type(8)))  float  v8f;
typedef __attribute__((ext_vector_type(4)))  unsigned int v4u;
typedef __attribute__((ext_vector_type(4)))  int  v4i;
typedef __attribute__((ext_vector_type(8)))  int  v8i;

// TDM builtin only exists in the device pass; the host pass still parses
// __device__ bodies, so all uses are guarded with __HIP_DEVICE_COMPILE__.
#if defined(__HIP_DEVICE_COMPILE__)
#if !__has_builtin(__builtin_amdgcn_tensor_load_to_lds)
#error "tensor_load_to_lds builtin not available on this device toolchain"
#endif
#endif

// ---------------- LDS layout (bytes) for the fused expert kernel ----------------
// XT  : 32 x 1024 bf16  x tile (row-major)                    =  65536
// WG0/1, WU0/1 : 64 x 64 bf16 gate/up chunks [i][h], dbl-buf  = 4x8192
// HH  : 32 x 64  bf16  silu(g)*u tile (row-major)             =   4096
// WD  : 1024 x 64 bf16 down-weight chunk [h][i]               = 131072
// SC  : 32 floats  per-token scale ; FL: routed flag
#define XT_OFF 0
#define WG0_OFF 65536
#define WG1_OFF (WG0_OFF + 8192)
#define WU0_OFF (WG1_OFF + 8192)
#define WU1_OFF (WU0_OFF + 8192)
#define HH_OFF  (WU1_OFF + 8192)
#define WD_OFF  (HH_OFF + 4096)
#define SC_OFF  (WD_OFF + 131072)
#define FL_OFF  (SC_OFF + 128)
#define SMEM_BYTES (FL_OFF + 16)

// fp32 -> bf16 round-to-nearest-even
__device__ __forceinline__ unsigned short f2bf(float f) {
  unsigned u = __builtin_bit_cast(unsigned, f);
  unsigned r = u + 0x7FFFu + ((u >> 16) & 1u);
  return (unsigned short)(r >> 16);
}

// Tensor Data Mover: async 2-D bf16 tile copy global -> LDS (D# per ISA ch.8).
// tensor: [tensor_d1 rows][tensor_d0 elems], row stride stride0 (elems).
// tile:   [tile_d1 rows][tile_d0 elems] starting at gsrc; lands contiguously in LDS.
__device__ __forceinline__ void tdm_load_2d(const unsigned short* lds_dst,
                                            const unsigned short* gsrc,
                                            unsigned tensor_d0, unsigned tensor_d1,
                                            unsigned tile_d0, unsigned tile_d1,
                                            unsigned long long stride0) {
#if defined(__HIP_DEVICE_COMPILE__)
  unsigned lds_addr = (unsigned)(unsigned long long)(uintptr_t)lds_dst; // low 32b = LDS offset
  unsigned long long ga = (unsigned long long)(uintptr_t)gsrc;
  v4u g0;
  g0.x = 1u;                                                   // count=1 (valid user D#)
  g0.y = lds_addr;                                             // D#.lds_addr
  g0.z = (unsigned)(ga & 0xffffffffull);                       // global_addr[31:0]
  g0.w = (unsigned)((ga >> 32) & 0x1ffffffull) | (2u << 30);   // addr[56:32] | type=2
  v8i g1;
  g1[0] = (int)(1u << 16);                                     // data_size=1 (2 bytes)
  g1[1] = (int)((tensor_d0 & 0xffffu) << 16);                  // tensor_dim0 lo
  g1[2] = (int)(((tensor_d0 >> 16) & 0xffffu) | ((tensor_d1 & 0xffffu) << 16));
  g1[3] = (int)(((tensor_d1 >> 16) & 0xffffu) | ((tile_d0 & 0xffffu) << 16));
  g1[4] = (int)(tile_d1 & 0xffffu);                            // tile_dim1 (tile_dim2=0)
  g1[5] = (int)(unsigned)(stride0 & 0xffffffffull);            // tensor_dim0_stride lo
  g1[6] = (int)(unsigned)((stride0 >> 32) & 0xffffull);        // stride hi (stride1=0)
  g1[7] = 0;
  v4i g2 = {0, 0, 0, 0};
  v4i g3 = {0, 0, 0, 0};
#if __clang_major__ >= 23
  v8i g4 = {0, 0, 0, 0, 0, 0, 0, 0};
  __builtin_amdgcn_tensor_load_to_lds(g0, g1, g2, g3, g4, 0);
#else
  __builtin_amdgcn_tensor_load_to_lds(g0, g1, g2, g3, 0);
#endif
#else
  (void)lds_dst; (void)gsrc; (void)tensor_d0; (void)tensor_d1;
  (void)tile_d0; (void)tile_d1; (void)stride0;
#endif
}

__device__ __forceinline__ void wait_tensorcnt0() {
#if defined(__HIP_DEVICE_COMPILE__)
  __builtin_amdgcn_s_wait_tensorcnt(0);
#endif
}

// WMMA A fragment (16x32 bf16) from row-major LDS tile.
__device__ __forceinline__ v16bf load_a(const unsigned short* p0, int mbase, int k0,
                                        int stride, int lane) {
  int m  = mbase + (lane & 15);
  int kb = k0 + ((lane >> 4) << 3);
  const unsigned short* p = p0 + m * stride + kb;
  union { uint4 q[2]; v16bf v; } un;
  un.q[0] = *(const uint4*)(p);
  un.q[1] = *(const uint4*)(p + 16);
  return un.v;
}

// WMMA B fragment (32x16 bf16) from [N][K] LDS tile.
__device__ __forceinline__ v16bf load_b(const unsigned short* p0, int nbase, int k0,
                                        int stride, int lane) {
  int n  = nbase + (lane & 15);
  int kb = k0 + ((lane >> 4) << 3);
  const unsigned short* p = p0 + n * stride + kb;
  union { uint4 q[2]; v16bf v; } un;
  un.q[0] = *(const uint4*)(p);
  un.q[1] = *(const uint4*)(p + 16);
  return un.v;
}

// -------------------------------------------------------------------------------
__global__ void zero_out_kernel(float* __restrict__ p, size_t n) {
  size_t i = (size_t)blockIdx.x * blockDim.x + threadIdx.x;
  size_t stride = (size_t)gridDim.x * blockDim.x;
  for (; i < n; i += stride) p[i] = 0.f;
}

// fp32 -> bf16 elementwise (for x)
__global__ void convert_bf16_kernel(const float* __restrict__ in,
                                    unsigned short* __restrict__ out, size_t n) {
  size_t i = (size_t)blockIdx.x * blockDim.x + threadIdx.x;
  size_t stride = (size_t)gridDim.x * blockDim.x;
  for (; i < n; i += stride) out[i] = f2bf(in[i]);
}

// Tiled transpose + convert: in [R][C] fp32 -> out [C][R] bf16. grid=(C/64, R/64, nmat)
__global__ void transpose_bf16_kernel(const float* __restrict__ in,
                                      unsigned short* __restrict__ out, int R, int C) {
  __shared__ unsigned short tile[64][65];
  const float* src = in + (size_t)blockIdx.z * R * C;
  unsigned short* dst = out + (size_t)blockIdx.z * R * C;
  int c0 = blockIdx.x * 64, r0 = blockIdx.y * 64;
  for (int idx = threadIdx.x; idx < 64 * 64; idx += 256) {
    int r = idx >> 6, c = idx & 63;
    tile[c][r] = f2bf(src[(size_t)(r0 + r) * C + c0 + c]);
  }
  __syncthreads();
  for (int idx = threadIdx.x; idx < 64 * 64; idx += 256) {
    int c = idx >> 6, r = idx & 63;
    dst[(size_t)(c0 + c) * R + r0 + r] = tile[c][r];
  }
}

// Router: one wave32 per token. 8 expert logits + shared sigmoid gate.
__global__ void moe_router_kernel(const float* __restrict__ x,
                                  const float* __restrict__ gate_w,
                                  const float* __restrict__ sgate_w,
                                  float* __restrict__ combine,
                                  float* __restrict__ sg) {
  int wave = threadIdx.x >> 5;
  int lane = threadIdx.x & 31;
  int t = blockIdx.x * 8 + wave;
  const float* xp = x + (size_t)t * H_DIM;

  float acc[E_NUM + 1];
#pragma unroll
  for (int e = 0; e <= E_NUM; ++e) acc[e] = 0.f;
  for (int h = lane; h < H_DIM; h += 32) {
    float xv = xp[h];
#pragma unroll
    for (int e = 0; e < E_NUM; ++e) acc[e] += xv * gate_w[e * H_DIM + h];
    acc[E_NUM] += xv * sgate_w[h];
  }
#pragma unroll
  for (int e = 0; e <= E_NUM; ++e) {
#pragma unroll
    for (int off = 16; off > 0; off >>= 1) acc[e] += __shfl_xor(acc[e], off, 32);
  }
  if (lane == 0) {
    float mx = acc[0];
#pragma unroll
    for (int e = 1; e < E_NUM; ++e) mx = fmaxf(mx, acc[e]);
    float p[E_NUM]; float sum = 0.f;
#pragma unroll
    for (int e = 0; e < E_NUM; ++e) { p[e] = __expf(acc[e] - mx); sum += p[e]; }
    float inv = 1.f / sum;
#pragma unroll
    for (int e = 0; e < E_NUM; ++e) p[e] *= inv;
    int i1 = 0; float w1 = p[0];
#pragma unroll
    for (int e = 1; e < E_NUM; ++e) if (p[e] > w1) { w1 = p[e]; i1 = e; }
    int i2 = (i1 == 0) ? 1 : 0; float w2 = p[i2];
#pragma unroll
    for (int e = 0; e < E_NUM; ++e) if (e != i1 && p[e] > w2) { w2 = p[e]; i2 = e; }
    float rn = 1.f / (w1 + w2);
    float* cb = combine + (size_t)t * E_NUM;
#pragma unroll
    for (int e = 0; e < E_NUM; ++e) cb[e] = 0.f;
    cb[i1] = w1 * rn;
    cb[i2] = w2 * rn;
    sg[t] = 1.f / (1.f + __expf(-acc[E_NUM]));
  }
}

// Fused SwiGLU expert kernel. grid=(T/32, E+1); block=256 (8 wave32).
// All LDS staging via Tensor Data Mover; wave0=WG, wave1=WU, wave2=WD, wave3=XT issuers.
__global__ void __launch_bounds__(256, 1) moe_experts_kernel(
    const unsigned short* __restrict__ xb,                       // [T][H] bf16
    const unsigned short* __restrict__ wgT,                      // [E][I][H] bf16
    const unsigned short* __restrict__ wuT,                      // [E][I][H] bf16
    const unsigned short* __restrict__ wdT,                      // [E][H][I] bf16
    const unsigned short* __restrict__ swgT,                     // [I][H] bf16
    const unsigned short* __restrict__ swuT,                     // [I][H] bf16
    const unsigned short* __restrict__ swdT,                     // [H][I] bf16
    const float* __restrict__ combine, const float* __restrict__ sg,
    float* __restrict__ out) {
  extern __shared__ char smem[];
  unsigned short* XT  = (unsigned short*)(smem + XT_OFF);
  unsigned short* WG0 = (unsigned short*)(smem + WG0_OFF);
  unsigned short* WG1 = (unsigned short*)(smem + WG1_OFF);
  unsigned short* WU0 = (unsigned short*)(smem + WU0_OFF);
  unsigned short* WU1 = (unsigned short*)(smem + WU1_OFF);
  unsigned short* HH  = (unsigned short*)(smem + HH_OFF);
  unsigned short* WD  = (unsigned short*)(smem + WD_OFF);
  float* SC = (float*)(smem + SC_OFF);
  int* FLAG = (int*)(smem + FL_OFF);

  const int tid  = threadIdx.x;
  const int lane = tid & 31;
  const int wave = tid >> 5;
  const int half = lane >> 4;
  const int t0 = blockIdx.x * 32;
  const int e  = blockIdx.y;

  const unsigned short* Wg = (e < E_NUM) ? (wgT + (size_t)e * I_DIM * H_DIM) : swgT;
  const unsigned short* Wu = (e < E_NUM) ? (wuT + (size_t)e * I_DIM * H_DIM) : swuT;
  const unsigned short* Wd = (e < E_NUM) ? (wdT + (size_t)e * H_DIM * I_DIM) : swdT;

  if (tid < 32)
    SC[tid] = (e < E_NUM) ? combine[(size_t)(t0 + tid) * E_NUM + e] : sg[t0 + tid];
  if (tid == 0) FLAG[0] = 0;
  __syncthreads();
  if (tid < 32 && SC[tid] != 0.f) atomicOr(FLAG, 1);
  __syncthreads();
  if (FLAG[0] == 0) return;  // no token in this tile routed to this expert

  // Async x-tile stage: [32 tokens][1024] bf16 -> XT (wave3's TENSORcnt).
  if (wave == 3)
    tdm_load_2d(XT, xb + (size_t)t0 * H_DIM, H_DIM, T_TOKENS, H_DIM, 32, H_DIM);

  // Persistent down-projection accumulators: wave owns 128 H columns -> 16 v8f.
  v8f cacc[2][8];
#pragma unroll
  for (int mt = 0; mt < 2; ++mt)
#pragma unroll
    for (int nt = 0; nt < 8; ++nt)
#pragma unroll
      for (int r = 0; r < 8; ++r) cacc[mt][nt][r] = 0.f;

  const int mtile_gu = wave >> 2;
  const int ntile_gu = wave & 3;
  const int hbase    = wave * 128;

  for (int c = 0; c < I_DIM / 64; ++c) {
    const int i0 = c * 64;

    // Kick off this chunk's staged transfers (overlap with compute below):
    if (wave == 0) tdm_load_2d(WG0, Wg + (size_t)i0 * H_DIM, H_DIM, I_DIM, 64, 64, H_DIM);
    if (wave == 1) tdm_load_2d(WU0, Wu + (size_t)i0 * H_DIM, H_DIM, I_DIM, 64, 64, H_DIM);
    if (wave == 2) tdm_load_2d(WD,  Wd + i0, I_DIM, H_DIM, 64, H_DIM, I_DIM); // overlaps gate/up

    // ---- gate & up GEMM: [32 x 64] = XT[32 x 1024] @ W[1024 x 64], dbl-buffered ----
    v8f g, u;
#pragma unroll
    for (int r = 0; r < 8; ++r) { g[r] = 0.f; u[r] = 0.f; }

    for (int ks = 0; ks < H_DIM; ks += 64) {
      const int pb = (ks >> 6) & 1;
      if (wave == 0 || wave == 1 || (wave == 3 && ks == 0 && c == 0))
        wait_tensorcnt0();
      __syncthreads();  // current WG/WU buffer (and XT on first pass) now valid
      if (ks + 64 < H_DIM) {  // prefetch next k-chunk into the other buffer
        if (wave == 0)
          tdm_load_2d(pb ? WG0 : WG1, Wg + (size_t)i0 * H_DIM + (ks + 64), H_DIM, I_DIM, 64, 64, H_DIM);
        if (wave == 1)
          tdm_load_2d(pb ? WU0 : WU1, Wu + (size_t)i0 * H_DIM + (ks + 64), H_DIM, I_DIM, 64, 64, H_DIM);
      }
      const unsigned short* WGc = pb ? WG1 : WG0;
      const unsigned short* WUc = pb ? WU1 : WU0;
#pragma unroll
      for (int kf = 0; kf < 2; ++kf) {
        v16bf a  = load_a(XT, mtile_gu * 16, ks + kf * 32, H_DIM, lane);
        v16bf bg = load_b(WGc, ntile_gu * 16, kf * 32, 64, lane);
        v16bf bu = load_b(WUc, ntile_gu * 16, kf * 32, 64, lane);
        g = __builtin_amdgcn_wmma_f32_16x16x32_bf16(false, a, false, bg, (short)0, g, false, false);
        u = __builtin_amdgcn_wmma_f32_16x16x32_bf16(false, a, false, bu, (short)0, u, false, false);
      }
    }

    // ---- hh = silu(g) * u -> LDS [32 x 64] bf16 ----
#pragma unroll
    for (int r = 0; r < 8; ++r) {
      float gv = g[r];
      float hv = (gv / (1.f + __expf(-gv))) * u[r];
      int m = mtile_gu * 16 + half * 8 + r;
      int n = ntile_gu * 16 + (lane & 15);
      HH[m * 64 + n] = f2bf(hv);
    }
    if (wave == 2) wait_tensorcnt0();  // WD chunk done
    __syncthreads();                   // HH visible + WD ready

    // ---- down GEMM: cacc[32 x 128/wave] += HH[32 x 64] @ WD[64 x 1024] ----
#pragma unroll
    for (int kf = 0; kf < 2; ++kf) {
      v16bf a0 = load_a(HH, 0,  kf * 32, 64, lane);
      v16bf a1 = load_a(HH, 16, kf * 32, 64, lane);
#pragma unroll
      for (int nt = 0; nt < 8; ++nt) {
        v16bf b = load_b(WD, hbase + nt * 16, kf * 32, 64, lane);
        cacc[0][nt] = __builtin_amdgcn_wmma_f32_16x16x32_bf16(false, a0, false, b, (short)0, cacc[0][nt], false, false);
        cacc[1][nt] = __builtin_amdgcn_wmma_f32_16x16x32_bf16(false, a1, false, b, (short)0, cacc[1][nt], false, false);
      }
    }
    __syncthreads();  // protect HH/WD before next chunk rewrites them
  }

  // ---- scale by per-token weight and accumulate into out ----
#pragma unroll
  for (int mt = 0; mt < 2; ++mt) {
#pragma unroll
    for (int nt = 0; nt < 8; ++nt) {
#pragma unroll
      for (int r = 0; r < 8; ++r) {
        int m = mt * 16 + half * 8 + r;
        int hcol = hbase + nt * 16 + (lane & 15);
        float v = SC[m] * cacc[mt][nt][r];
        unsafeAtomicAdd(&out[(size_t)(t0 + m) * H_DIM + hcol], v);
      }
    }
  }
}

// -------------------------------------------------------------------------------
extern "C" void kernel_launch(void* const* d_in, const int* in_sizes, int n_in,
                              void* d_out, int out_size, void* d_ws, size_t ws_size,
                              hipStream_t stream) {
  const float* x        = (const float*)d_in[0];  // [B,S,H]
  const float* gate_w   = (const float*)d_in[1];  // [E,H]
  const float* w_gate   = (const float*)d_in[2];  // [E,H,I]
  const float* w_up     = (const float*)d_in[3];  // [E,H,I]
  const float* w_down   = (const float*)d_in[4];  // [E,I,H]
  const float* sw_gate  = (const float*)d_in[5];  // [H,I]
  const float* sw_up    = (const float*)d_in[6];  // [H,I]
  const float* sw_down  = (const float*)d_in[7];  // [I,H]
  const float* sgate_w  = (const float*)d_in[8];  // [1,H]
  float* out = (float*)d_out;

  // Workspace layout (all region sizes are 256B multiples).
  char* ws = (char*)d_ws;
  float* combine = (float*)ws;                 ws += (size_t)T_TOKENS * E_NUM * 4;
  float* sg      = (float*)ws;                 ws += (size_t)T_TOKENS * 4;
  unsigned short* xb   = (unsigned short*)ws;  ws += (size_t)T_TOKENS * H_DIM * 2;
  unsigned short* wgT  = (unsigned short*)ws;  ws += (size_t)E_NUM * I_DIM * H_DIM * 2;
  unsigned short* wuT  = (unsigned short*)ws;  ws += (size_t)E_NUM * I_DIM * H_DIM * 2;
  unsigned short* wdT  = (unsigned short*)ws;  ws += (size_t)E_NUM * H_DIM * I_DIM * 2;
  unsigned short* swgT = (unsigned short*)ws;  ws += (size_t)I_DIM * H_DIM * 2;
  unsigned short* swuT = (unsigned short*)ws;  ws += (size_t)I_DIM * H_DIM * 2;
  unsigned short* swdT = (unsigned short*)ws;  ws += (size_t)H_DIM * I_DIM * 2;
  (void)in_sizes; (void)n_in; (void)ws_size;

  (void)hipFuncSetAttribute((const void*)moe_experts_kernel,
                            hipFuncAttributeMaxDynamicSharedMemorySize, SMEM_BYTES);

  zero_out_kernel<<<4096, 256, 0, stream>>>(out, (size_t)out_size);
  convert_bf16_kernel<<<4096, 256, 0, stream>>>(x, xb, (size_t)T_TOKENS * H_DIM);
  // Weight pre-pass: convert + transpose into WMMA-friendly [N][K] bf16 layouts.
  transpose_bf16_kernel<<<dim3(I_DIM / 64, H_DIM / 64, E_NUM), 256, 0, stream>>>(w_gate, wgT, H_DIM, I_DIM);
  transpose_bf16_kernel<<<dim3(I_DIM / 64, H_DIM / 64, E_NUM), 256, 0, stream>>>(w_up,   wuT, H_DIM, I_DIM);
  transpose_bf16_kernel<<<dim3(H_DIM / 64, I_DIM / 64, E_NUM), 256, 0, stream>>>(w_down, wdT, I_DIM, H_DIM);
  transpose_bf16_kernel<<<dim3(I_DIM / 64, H_DIM / 64, 1), 256, 0, stream>>>(sw_gate, swgT, H_DIM, I_DIM);
  transpose_bf16_kernel<<<dim3(I_DIM / 64, H_DIM / 64, 1), 256, 0, stream>>>(sw_up,   swuT, H_DIM, I_DIM);
  transpose_bf16_kernel<<<dim3(H_DIM / 64, I_DIM / 64, 1), 256, 0, stream>>>(sw_down, swdT, I_DIM, H_DIM);

  moe_router_kernel<<<T_TOKENS / 8, 256, 0, stream>>>(x, gate_w, sgate_w, combine, sg);

  dim3 grid(T_TOKENS / 32, E_NUM + 1);
  moe_experts_kernel<<<grid, 256, SMEM_BYTES, stream>>>(
      xb, wgT, wuT, wdT, swgT, swuT, swdT, combine, sg, out);
}
